// TopKSAE_2808908612115
// MI455X (gfx1250) — compile-verified, matrix operand order checked by reading, not run on hardware
//
#include <hip/hip_runtime.h>

#define BATCH 4096
#define D_IN  1024
#define HID   16384
#define KSEL  32

typedef float v2f  __attribute__((ext_vector_type(2)));
typedef float v8f  __attribute__((ext_vector_type(8)));
typedef __bf16 bf16_t;
typedef bf16_t v16bf __attribute__((ext_vector_type(16)));
typedef bf16_t v4bf  __attribute__((ext_vector_type(4)));
typedef unsigned int v4ui __attribute__((ext_vector_type(4)));
typedef int v4si __attribute__((ext_vector_type(4)));
typedef int v8si __attribute__((ext_vector_type(8)));

struct Pair { float v; int idx; };

#define BM 128
#define BN 128
#define BK 32

// ---------------------------------------------------------------------
// TDM: 2D tile load (tile_k=32 x tile_rows=128, bf16) global -> LDS.
// D# per CDNA5 ISA. Rows contiguous in LDS
// (row stride = 32 bf16 = 64B). Tracked on TENSORcnt.
// ---------------------------------------------------------------------
__device__ __forceinline__ void tdm_load_tile_bf16(const bf16_t* gptr,
                                                   unsigned lds_off,
                                                   unsigned tensor_rows)
{
    const unsigned long long ga = (unsigned long long)(uintptr_t)gptr;
    v4ui g0;
    g0[0] = 1u;                                      // count=1, user mode
    g0[1] = lds_off;                                 // lds_addr (bytes)
    g0[2] = (unsigned)(ga & 0xffffffffu);            // global_addr[31:0]
    g0[3] = (unsigned)((ga >> 32) & 0x01ffffffu)     // global_addr[56:32]
          | (2u << 30);                              // type=2 ("image")
    v8si g1;
    g1[0] = (int)(1u << 16);                         // data_size=1 (2 bytes)
    g1[1] = (int)(((unsigned)D_IN & 0xffffu) << 16); // tensor_dim0 lo16 (=1024)
    g1[2] = (int)((((unsigned)D_IN >> 16) & 0xffffu)
          | ((tensor_rows & 0xffffu) << 16));        // dim0 hi16 | tensor_dim1 lo16
    g1[3] = (int)(((tensor_rows >> 16) & 0xffffu)
          | ((unsigned)BK << 16));                   // tensor_dim1 hi16 | tile_dim0=32
    g1[4] = (int)(unsigned)BM;                       // tile_dim1=128, tile_dim2=0
    g1[5] = (int)(unsigned)D_IN;                     // tensor_dim0_stride lo32
    g1[6] = 0;                                       // stride hi | dim1_stride lo
    g1[7] = 0;
    const v4si z4 = {0, 0, 0, 0};
#if defined(__clang_major__) && (__clang_major__ >= 23)
    const v8si z8 = {0, 0, 0, 0, 0, 0, 0, 0};
    __builtin_amdgcn_tensor_load_to_lds(g0, g1, z4, z4, z8, 0);
#else
    __builtin_amdgcn_tensor_load_to_lds(g0, g1, z4, z4, 0);
#endif
}

// ---------------------------------------------------------------------
// Kernel 0: split f32 -> (hi, lo) bf16  (one streaming pass)
// ---------------------------------------------------------------------
__global__ __launch_bounds__(256)
void sae_split_bf16(const float* __restrict__ src,
                    bf16_t* __restrict__ hi, bf16_t* __restrict__ lo, int n4)
{
    const int i = blockIdx.x * 256 + threadIdx.x;
    if (i >= n4) return;
    const float4 v = ((const float4*)src)[i];
    v4bf hv, lv;
    float f;
    f = v.x; hv[0] = (bf16_t)f; lv[0] = (bf16_t)(f - (float)hv[0]);
    f = v.y; hv[1] = (bf16_t)f; lv[1] = (bf16_t)(f - (float)hv[1]);
    f = v.z; hv[2] = (bf16_t)f; lv[2] = (bf16_t)(f - (float)hv[2]);
    f = v.w; hv[3] = (bf16_t)f; lv[3] = (bf16_t)(f - (float)hv[3]);
    *(v4bf*)(hi + 4 * (size_t)i) = hv;
    *(v4bf*)(lo + 4 * (size_t)i) = lv;
}

// ---------------------------------------------------------------------
// Kernel 1a (primary): encoder GEMM, bf16x3 split product, f32 accum.
// enc = x @ Wenc^T + benc.  TDM double-buffered LDS staging.
// 128x128 block, 8 wave32 (4x2), wave tile 32x64, 24 WMMA / K=32 stage.
// ---------------------------------------------------------------------
__global__ __launch_bounds__(256)
void sae_encoder_gemm_bf16x3(const bf16_t* __restrict__ xh, const bf16_t* __restrict__ xl,
                             const bf16_t* __restrict__ Wh, const bf16_t* __restrict__ Wl,
                             const float* __restrict__ benc, float* __restrict__ enc)
{
    __shared__ bf16_t Ah[2][BM * BK];
    __shared__ bf16_t Al[2][BM * BK];
    __shared__ bf16_t Bh[2][BN * BK];
    __shared__ bf16_t Bl[2][BN * BK];

    const int tid  = threadIdx.x;
    const int lane = tid & 31;
    const int wid  = tid >> 5;
    const int wm   = wid >> 1;       // 0..3
    const int wn   = wid & 1;        // 0..1
    const int l    = lane & 15;
    const int h    = lane >> 4;
    const int m_base = blockIdx.y * BM;
    const int n_base = blockIdx.x * BN;

    v8f acc[2][4];
#pragma unroll
    for (int i = 0; i < 2; ++i)
#pragma unroll
        for (int j = 0; j < 4; ++j)
#pragma unroll
            for (int e = 0; e < 8; ++e)
                acc[i][j][e] = 0.0f;

    auto issue = [&](int k0, int buf) {
        tdm_load_tile_bf16(xh + (size_t)m_base * D_IN + k0,
                           (unsigned)(uintptr_t)&Ah[buf][0], BATCH);
        tdm_load_tile_bf16(xl + (size_t)m_base * D_IN + k0,
                           (unsigned)(uintptr_t)&Al[buf][0], BATCH);
        tdm_load_tile_bf16(Wh + (size_t)n_base * D_IN + k0,
                           (unsigned)(uintptr_t)&Bh[buf][0], HID);
        tdm_load_tile_bf16(Wl + (size_t)n_base * D_IN + k0,
                           (unsigned)(uintptr_t)&Bl[buf][0], HID);
    };

    if (tid == 0) {
        issue(0, 0);
        __builtin_amdgcn_s_wait_tensorcnt(0);
    }
    __syncthreads();

    int buf = 0;
    for (int k0 = 0; k0 < D_IN; k0 += BK) {
        const bool more = (k0 + BK) < D_IN;
        if (tid == 0 && more) issue(k0 + BK, buf ^ 1);   // overlap DMA w/ compute

        // A fragments (16-bit A 16x32 layout: elems 0..7 = K h*8.., 8..15 = K 16+h*8..)
        v16bf ahf[2], alf[2];
#pragma unroll
        for (int i = 0; i < 2; ++i) {
            const bf16_t* ph = &Ah[buf][(wm * 32 + i * 16 + l) * BK + h * 8];
            const bf16_t* pl = &Al[buf][(wm * 32 + i * 16 + l) * BK + h * 8];
#pragma unroll
            for (int q = 0; q < 8; ++q) {
                ahf[i][q] = ph[q];      ahf[i][8 + q] = ph[16 + q];
                alf[i][q] = pl[q];      alf[i][8 + q] = pl[16 + q];
            }
        }
#pragma unroll
        for (int j = 0; j < 4; ++j) {
            v16bf bhf, blf;
            const bf16_t* ph = &Bh[buf][(wn * 64 + j * 16 + l) * BK + h * 8];
            const bf16_t* pl = &Bl[buf][(wn * 64 + j * 16 + l) * BK + h * 8];
#pragma unroll
            for (int q = 0; q < 8; ++q) {
                bhf[q] = ph[q];         bhf[8 + q] = ph[16 + q];
                blf[q] = pl[q];         blf[8 + q] = pl[16 + q];
            }
#pragma unroll
            for (int i = 0; i < 2; ++i) {
                acc[i][j] = __builtin_amdgcn_wmma_f32_16x16x32_bf16(
                    false, ahf[i], false, bhf, (short)0, acc[i][j], false, false);
                acc[i][j] = __builtin_amdgcn_wmma_f32_16x16x32_bf16(
                    false, ahf[i], false, blf, (short)0, acc[i][j], false, false);
                acc[i][j] = __builtin_amdgcn_wmma_f32_16x16x32_bf16(
                    false, alf[i], false, bhf, (short)0, acc[i][j], false, false);
            }
        }

        __syncthreads();                     // everyone done reading buf
        if (tid == 0 && more) __builtin_amdgcn_s_wait_tensorcnt(0);
        __syncthreads();                     // publish next buffer
        buf ^= 1;
    }

#pragma unroll
    for (int i = 0; i < 2; ++i) {
#pragma unroll
        for (int j = 0; j < 4; ++j) {
            const int n = n_base + wn * 64 + j * 16 + l;
            const float bias = benc[n];
#pragma unroll
            for (int e = 0; e < 8; ++e) {
                const int m = m_base + wm * 32 + i * 16 + e + 8 * h;
                enc[(size_t)m * HID + n] = acc[i][j][e] + bias;
            }
        }
    }
}

// ---------------------------------------------------------------------
// Kernel 1b (fallback, exact f32): WMMA 16x16x4 f32 GEMM (as round 1).
// ---------------------------------------------------------------------
__global__ __launch_bounds__(256)
void sae_encoder_gemm_f32(const float* __restrict__ x,
                          const float* __restrict__ Wenc,
                          const float* __restrict__ benc,
                          float* __restrict__ enc)
{
    __shared__ float As[BM][BK + 1];
    __shared__ float Bs[BN][BK + 1];

    const int tid  = threadIdx.x;
    const int lane = tid & 31;
    const int wid  = tid >> 5;
    const int wm   = wid >> 1;
    const int wn   = wid & 1;
    const int l    = lane & 15;
    const int h    = lane >> 4;
    const int m_base = blockIdx.y * BM;
    const int n_base = blockIdx.x * BN;

    v8f acc[2][4];
#pragma unroll
    for (int i = 0; i < 2; ++i)
#pragma unroll
        for (int j = 0; j < 4; ++j)
#pragma unroll
            for (int e = 0; e < 8; ++e)
                acc[i][j][e] = 0.0f;

    for (int k0 = 0; k0 < D_IN; k0 += BK) {
#pragma unroll
        for (int it = 0; it < 4; ++it) {
            const int pos = tid + it * 256;
            const int row = pos >> 3;
            const int c4  = (pos & 7) << 2;
            const float4 va = *(const float4*)(x    + (size_t)(m_base + row) * D_IN + k0 + c4);
            const float4 vb = *(const float4*)(Wenc + (size_t)(n_base + row) * D_IN + k0 + c4);
            As[row][c4 + 0] = va.x; As[row][c4 + 1] = va.y;
            As[row][c4 + 2] = va.z; As[row][c4 + 3] = va.w;
            Bs[row][c4 + 0] = vb.x; Bs[row][c4 + 1] = vb.y;
            Bs[row][c4 + 2] = vb.z; Bs[row][c4 + 3] = vb.w;
        }
        __syncthreads();

        if (k0 + BK < D_IN) {
            const int prow = tid >> 1;
            const int pc   = (tid & 1) * 16;
            __builtin_prefetch(x    + (size_t)(m_base + prow) * D_IN + k0 + BK + pc, 0, 3);
            __builtin_prefetch(Wenc + (size_t)(n_base + prow) * D_IN + k0 + BK + pc, 0, 3);
        }

#pragma unroll
        for (int kk = 0; kk < BK; kk += 4) {
            v2f a[2], b[4];
#pragma unroll
            for (int i = 0; i < 2; ++i) {
                const float* ap = &As[wm * 32 + i * 16 + l][kk + 2 * h];
                a[i].x = ap[0]; a[i].y = ap[1];
            }
#pragma unroll
            for (int j = 0; j < 4; ++j) {
                const float* bp = &Bs[wn * 64 + j * 16 + l][kk + 2 * h];
                b[j].x = bp[0]; b[j].y = bp[1];
            }
#pragma unroll
            for (int i = 0; i < 2; ++i)
#pragma unroll
                for (int j = 0; j < 4; ++j)
                    acc[i][j] = __builtin_amdgcn_wmma_f32_16x16x4_f32(
                        false, a[i], false, b[j], (short)0, acc[i][j], false, false);
        }
        __syncthreads();
    }

#pragma unroll
    for (int i = 0; i < 2; ++i) {
#pragma unroll
        for (int j = 0; j < 4; ++j) {
            const int n = n_base + wn * 64 + j * 16 + l;
            const float bias = benc[n];
#pragma unroll
            for (int e = 0; e < 8; ++e) {
                const int m = m_base + wm * 32 + i * 16 + e + 8 * h;
                enc[(size_t)m * HID + n] = acc[i][j][e] + bias;
            }
        }
    }
}

// ---------------------------------------------------------------------
// Kernel 2: per-row exact top-32 by |value| (MSB-first radix select).
// Deterministic tie handling by index. Zeros non-kept in place.
// ---------------------------------------------------------------------
__global__ __launch_bounds__(256)
void sae_topk(float* __restrict__ enc, Pair* __restrict__ pairs)
{
    __shared__ unsigned int hist[256];
    __shared__ unsigned int sh_keep[256], sh_ties[256];
    __shared__ unsigned int tieBase[256], slotBase[256];
    __shared__ unsigned int s_prefix, s_remaining;

    const int row = blockIdx.x;
    const int tid = threadIdx.x;
    float* base = enc + (size_t)row * HID;
    const int start = tid * (HID / 256);

    if (tid == 0) { s_prefix = 0u; s_remaining = KSEL; }
    __syncthreads();

    for (int shift = 24; shift >= 0; shift -= 8) {
        hist[tid] = 0u;
        __syncthreads();
        const unsigned int pfx = s_prefix;
        for (int e = 0; e < HID / 256; ++e) {
            const unsigned int key = __float_as_uint(base[start + e]) & 0x7fffffffu;
            const bool match = (shift == 24) || ((key >> (shift + 8)) == pfx);
            if (match) atomicAdd(&hist[(key >> shift) & 255u], 1u);
        }
        __syncthreads();
        if (tid == 0) {
            const unsigned int rem = s_remaining;
            unsigned int cum = 0u;
            for (int b = 255; b >= 0; --b) {
                const unsigned int c = hist[b];
                if (cum + c >= rem) {
                    s_prefix = (s_prefix << 8) | (unsigned int)b;
                    s_remaining = rem - cum;
                    break;
                }
                cum += c;
            }
        }
        __syncthreads();
    }

    const unsigned int T = s_prefix;
    const unsigned int needed = s_remaining;

    unsigned int nk = 0, nt = 0;
    for (int e = 0; e < HID / 256; ++e) {
        const unsigned int key = __float_as_uint(base[start + e]) & 0x7fffffffu;
        if (key > T) nk++; else if (key == T) nt++;
    }
    sh_keep[tid] = nk; sh_ties[tid] = nt;
    __syncthreads();

    if (tid == 0) {
        unsigned int tacc = 0, sacc = 0;
        for (int t = 0; t < 256; ++t) {
            tieBase[t] = tacc;
            unsigned int takeT = 0;
            if (tacc < needed) {
                const unsigned int r = needed - tacc;
                takeT = (sh_ties[t] < r) ? sh_ties[t] : r;
            }
            slotBase[t] = sacc;
            sacc += sh_keep[t] + takeT;
            tacc += sh_ties[t];
        }
    }
    __syncthreads();

    unsigned int tcnt = tieBase[tid];
    unsigned int slot = slotBase[tid];
    for (int e = 0; e < HID / 256; ++e) {
        const int idx = start + e;
        const float v = base[idx];
        const unsigned int key = __float_as_uint(v) & 0x7fffffffu;
        const bool keep = (key > T) || ((key == T) && (tcnt < needed));
        if (key == T) tcnt++;
        if (keep) {
            Pair p; p.v = v; p.idx = idx;
            pairs[row * KSEL + slot] = p;
            slot++;
        } else {
            base[idx] = 0.0f;
        }
    }
}

// ---------------------------------------------------------------------
// Kernel 3: W_dec transpose [D_IN,HID] -> Wt [HID,D_IN]
// ---------------------------------------------------------------------
__global__ __launch_bounds__(256)
void sae_transpose_wdec(const float* __restrict__ W, float* __restrict__ Wt)
{
    __shared__ float t[32][33];
    const int hx = blockIdx.x * 32;
    const int dy = blockIdx.y * 32;
    const int lx = threadIdx.x & 31;
    const int ly = threadIdx.x >> 5;
#pragma unroll
    for (int r = 0; r < 4; ++r)
        t[ly + r * 8][lx] = W[(size_t)(dy + ly + r * 8) * HID + hx + lx];
    __syncthreads();
#pragma unroll
    for (int r = 0; r < 4; ++r)
        Wt[(size_t)(hx + ly + r * 8) * D_IN + dy + lx] = t[lx][ly + r * 8];
}

// ---------------------------------------------------------------------
// Kernel 4: sparse decode  dec[b,:] = sum_j v_j * Wt[idx_j,:] + b_dec
// ---------------------------------------------------------------------
__global__ __launch_bounds__(256)
void sae_decoder_t(const Pair* __restrict__ pairs, const float* __restrict__ Wt,
                   const float* __restrict__ bdec, float* __restrict__ dec)
{
    __shared__ Pair sp[KSEL];
    const int row = blockIdx.x;
    const int tid = threadIdx.x;
    if (tid < KSEL) sp[tid] = pairs[row * KSEL + tid];
    __syncthreads();

    const int d = tid * 4;
    float4 a = *(const float4*)(bdec + d);
#pragma unroll 4
    for (int j = 0; j < KSEL; ++j) {
        const float  v = sp[j].v;
        const float4 w = *(const float4*)(Wt + (size_t)sp[j].idx * D_IN + d);
        a.x = fmaf(v, w.x, a.x); a.y = fmaf(v, w.y, a.y);
        a.z = fmaf(v, w.z, a.z); a.w = fmaf(v, w.w, a.w);
    }
    *(float4*)(dec + (size_t)row * D_IN + d) = a;
}

__global__ __launch_bounds__(256)
void sae_decoder_fallback(const Pair* __restrict__ pairs, const float* __restrict__ Wdec,
                          const float* __restrict__ bdec, float* __restrict__ dec)
{
    __shared__ Pair sp[KSEL];
    const int row = blockIdx.x;
    const int tid = threadIdx.x;
    if (tid < KSEL) sp[tid] = pairs[row * KSEL + tid];
    __syncthreads();

    const int d = tid * 4;
    float a0 = bdec[d + 0], a1 = bdec[d + 1], a2 = bdec[d + 2], a3 = bdec[d + 3];
    for (int j = 0; j < KSEL; ++j) {
        const float v = sp[j].v;
        const size_t col = (size_t)sp[j].idx;
        a0 = fmaf(v, Wdec[(size_t)(d + 0) * HID + col], a0);
        a1 = fmaf(v, Wdec[(size_t)(d + 1) * HID + col], a1);
        a2 = fmaf(v, Wdec[(size_t)(d + 2) * HID + col], a2);
        a3 = fmaf(v, Wdec[(size_t)(d + 3) * HID + col], a3);
    }
    float* o = dec + (size_t)row * D_IN + d;
    o[0] = a0; o[1] = a1; o[2] = a2; o[3] = a3;
}

// =====================================================================
extern "C" void kernel_launch(void* const* d_in, const int* in_sizes, int n_in,
                              void* d_out, int out_size, void* d_ws, size_t ws_size,
                              hipStream_t stream)
{
    const float* x    = (const float*)d_in[0];
    const float* Wenc = (const float*)d_in[1];
    const float* benc = (const float*)d_in[2];
    const float* Wdec = (const float*)d_in[3];
    const float* bdec = (const float*)d_in[4];

    float* enc_out = (float*)d_out;                     // [BATCH, HID]
    float* dec_out = enc_out + (size_t)BATCH * HID;     // [BATCH, D_IN]

    // workspace layout (256B-aligned sections)
    size_t off = 0;
    auto take = [&](size_t bytes) {
        void* p = (char*)d_ws + off;
        off = (off + bytes + 255) & ~(size_t)255;
        return p;
    };
    Pair*   pairs = (Pair*)  take((size_t)BATCH * KSEL * sizeof(Pair));
    const size_t off_pairs = off;
    float*  Wt    = (float*) take((size_t)HID * D_IN * sizeof(float));
    const size_t off_wt = off;
    bf16_t* xh    = (bf16_t*)take((size_t)BATCH * D_IN * sizeof(bf16_t));
    bf16_t* xl    = (bf16_t*)take((size_t)BATCH * D_IN * sizeof(bf16_t));
    bf16_t* Wh    = (bf16_t*)take((size_t)HID * D_IN * sizeof(bf16_t));
    bf16_t* Wl    = (bf16_t*)take((size_t)HID * D_IN * sizeof(bf16_t));
    const size_t off_split = off;

    const bool useT     = ws_size >= off_wt;     // transposed decode path
    const bool useSplit = ws_size >= off_split;  // bf16x3 + TDM encoder path

    if (useSplit) {
        const int nx4 = (BATCH * D_IN) / 4;
        const int nw4 = (HID * D_IN) / 4;
        sae_split_bf16<<<(nx4 + 255) / 256, 256, 0, stream>>>(x, xh, xl, nx4);
        sae_split_bf16<<<(nw4 + 255) / 256, 256, 0, stream>>>(Wenc, Wh, Wl, nw4);
        sae_encoder_gemm_bf16x3<<<dim3(HID / BN, BATCH / BM), 256, 0, stream>>>(
            xh, xl, Wh, Wl, benc, enc_out);
    } else {
        sae_encoder_gemm_f32<<<dim3(HID / BN, BATCH / BM), 256, 0, stream>>>(
            x, Wenc, benc, enc_out);
    }

    if (useT)
        sae_transpose_wdec<<<dim3(HID / 32, D_IN / 32), 256, 0, stream>>>(Wdec, Wt);

    sae_topk<<<BATCH, 256, 0, stream>>>(enc_out, pairs);

    if (useT)
        sae_decoder_t<<<BATCH, 256, 0, stream>>>(pairs, Wt, bdec, dec_out);
    else
        sae_decoder_fallback<<<BATCH, 256, 0, stream>>>(pairs, Wdec, bdec, dec_out);
}
